// StringLabelEncoder_20366734917919
// MI455X (gfx1250) — 8-bit matrix operand order repaired and checked by reading, compiled, not run on hardware
//
#include <hip/hip_runtime.h>

typedef __attribute__((ext_vector_type(8))) int v8i;

#define N_ROWS    4096
#define N_CLASSES 50000
#define CHUNK_CLS 400
#define CHUNK_TILES (CHUNK_CLS / 16)
#define N_CHUNKS  (N_CLASSES / CHUNK_CLS)

__device__ __forceinline__ int ssq_u8x4(unsigned w) {
    unsigned b0 = w & 255u, b1 = (w >> 8) & 255u, b2 = (w >> 16) & 255u, b3 = (w >> 24) & 255u;
    return (int)(b0 * b0 + b1 * b1 + b2 * b2 + b3 * b3);
}

__device__ __forceinline__ v8i load_B(const int4* s_cls, int idx) {
    int4 c = s_cls[idx];
    v8i B;
    const bool bhi = (threadIdx.x & 31) >= 16;
    B[0] = bhi ? c.z : c.x; B[1] = bhi ? c.w : c.y; B[2] = 0; B[3] = 0;
    B[4] = 0;   B[5] = 0;   B[6] = 0;   B[7] = 0;
    return B;
}

__global__ __launch_bounds__(256) void
StringLabelEncoder_match_kernel(const int* __restrict__ x,
                                const int* __restrict__ classes,
                                float* __restrict__ out) {
    __shared__ int4 s_cls[CHUNK_CLS];
    __shared__ int  s_nb[CHUNK_CLS];

    const int lane = threadIdx.x & 31;
    const int wave = threadIdx.x >> 5;
    const int tile = blockIdx.x * 8 + wave;
    const int rowBase = tile * 16;
    const bool hi = lane >= 16;
    const int r = lane & 15;

    const int4 xr = ((const int4*)x)[rowBase + r];
    v8i A = (v8i)0;
    A[0] = hi ? xr.z : xr.x;
    A[1] = hi ? xr.w : xr.y;

    int partA = ssq_u8x4((unsigned)A[0]) + ssq_u8x4((unsigned)A[1]);
    int normArow = partA + __shfl_xor(partA, 16, 32);

    int normAv[8];
#pragma unroll
    for (int v = 0; v < 8; ++v)
        normAv[v] = __shfl(normArow, hi ? (v + 8) : v, 32);

    int acc[8];
#pragma unroll
    for (int v = 0; v < 8; ++v) acc[v] = 0;

    for (int chunk = 0; chunk < N_CHUNKS; ++chunk) {
        const int clsBase = chunk * CHUNK_CLS;

        __syncthreads();
        for (int i = threadIdx.x; i < CHUNK_CLS; i += 256) {
            int4 c = ((const int4*)classes)[clsBase + i];
            s_cls[i] = c;
            s_nb[i] = ssq_u8x4((unsigned)c.x) + ssq_u8x4((unsigned)c.y) +
                      ssq_u8x4((unsigned)c.z) + ssq_u8x4((unsigned)c.w);
        }
        __syncthreads();

        v8i B0 = load_B(s_cls, r);
        int nb0 = s_nb[r];
        v8i D0 = __builtin_amdgcn_wmma_i32_16x16x64_iu8(
            false, A, false, B0, (v8i)0, false, false);

        v8i B1 = load_B(s_cls, 16 + r);
        int nb1 = s_nb[16 + r];
        v8i D1 = __builtin_amdgcn_wmma_i32_16x16x64_iu8(
            false, A, false, B1, (v8i)0, false, false);

#pragma unroll
        for (int t = 2; t < CHUNK_TILES; ++t) {
            v8i Bn = load_B(s_cls, t * 16 + r);
            int nbn = s_nb[t * 16 + r];
            v8i Dn = __builtin_amdgcn_wmma_i32_16x16x64_iu8(
                false, A, false, Bn, (v8i)0, false, false);

            const int jr = clsBase + (t - 2) * 16 + r;
#pragma unroll
            for (int v = 0; v < 8; ++v) {
                bool m = (D0[v] + D0[v]) == (normAv[v] + nb0);
                acc[v] = m ? jr : acc[v];
            }
            D0 = D1; nb0 = nb1;
            D1 = Dn; nb1 = nbn;
        }

        {
            const int jr = clsBase + (CHUNK_TILES - 2) * 16 + r;
#pragma unroll
            for (int v = 0; v < 8; ++v) {
                bool m = (D0[v] + D0[v]) == (normAv[v] + nb0);
                acc[v] = m ? jr : acc[v];
            }
        }
        {
            const int jr = clsBase + (CHUNK_TILES - 1) * 16 + r;
#pragma unroll
            for (int v = 0; v < 8; ++v) {
                bool m = (D1[v] + D1[v]) == (normAv[v] + nb1);
                acc[v] = m ? jr : acc[v];
            }
        }
    }

#pragma unroll
    for (int v = 0; v < 8; ++v) {
        acc[v] += __shfl_xor(acc[v], 1, 32);
        acc[v] += __shfl_xor(acc[v], 2, 32);
        acc[v] += __shfl_xor(acc[v], 4, 32);
        acc[v] += __shfl_xor(acc[v], 8, 32);
    }

    if (r == 0) {
        const int base = rowBase + (hi ? 8 : 0);
#pragma unroll
        for (int v = 0; v < 8; ++v)
            out[base + v] = (float)acc[v];
    }
}

extern "C" void kernel_launch(void* const* d_in, const int* in_sizes, int n_in,
                              void* d_out, int out_size, void* d_ws, size_t ws_size,
                              hipStream_t stream) {
    (void)in_sizes; (void)n_in; (void)d_ws; (void)ws_size; (void)out_size;
    const int* x       = (const int*)d_in[0];
    const int* classes = (const int*)d_in[1];
    float* out = (float*)d_out;

    StringLabelEncoder_match_kernel<<<32, 256, 0, stream>>>(x, classes, out);
}
